// NeighborAttentionLayer_52501680226798
// MI455X (gfx1250) — compile-verified, matrix operand order checked by reading, not run on hardware
//
#include <hip/hip_runtime.h>
#include <hip/hip_bf16.h>
#include <math.h>

// N=8192, F=128, DK=32, KN=16, EMB=64
#define F_    128
#define DK_   32
#define KN_   16
#define EMB_  64
#define ALPHA_ 0.2f
#define INV_SCALE 0.17677669529663687f   // 1/sqrt(32)

typedef __attribute__((ext_vector_type(16))) __bf16 v16bf;
typedef __attribute__((ext_vector_type(8)))  __bf16 v8bf;
typedef __attribute__((ext_vector_type(8)))  float  v8f;
typedef __attribute__((ext_vector_type(4)))  unsigned int v4u;
typedef __attribute__((ext_vector_type(8)))  int  v8i;
typedef __attribute__((ext_vector_type(4)))  int  v4i;

#if __has_builtin(__builtin_amdgcn_tensor_load_to_lds) && \
    __has_builtin(__builtin_amdgcn_s_wait_tensorcnt)
#define USE_TDM 1
#else
#define USE_TDM 0
#endif

// LDS byte offset of a __shared__ object (ptrtoint of the as(3) pointer).
__device__ __forceinline__ unsigned lds_offset(const void* p) {
    return (unsigned)(size_t)(__attribute__((address_space(3))) const char*)p;
}

// Load a WMMA A/B fragment (16 rows x K=32, bf16) from a row-major
// [nrows][32] bf16 LDS array. Per CDNA5 ISA 16-bit A layout:
//   lanes 0-15  hold row l,  K = 0..7   and 16..23
//   lanes 16-31 hold row l,  K = 8..15  and 24..31
__device__ __forceinline__ v16bf ld_frag32(const __bf16* base, int row0, int lane) {
    const int l  = lane & 15;
    const int hi = (lane >> 4) << 3;               // 0 or 8
    const __bf16* r = base + (row0 + l) * DK_;
    v8bf a = *(const v8bf*)(r + hi);               // K = hi .. hi+7
    v8bf b = *(const v8bf*)(r + 16 + hi);          // K = 16+hi .. 16+hi+7
    return __builtin_shufflevector(a, b, 0,1,2,3,4,5,6,7,8,9,10,11,12,13,14,15);
}

__device__ __forceinline__ float wave_max(float v) {
#pragma unroll
    for (int off = 16; off > 0; off >>= 1) v = fmaxf(v, __shfl_xor(v, off, 32));
    return v;
}
__device__ __forceinline__ float wave_sum(float v) {
#pragma unroll
    for (int off = 16; off > 0; off >>= 1) v += __shfl_xor(v, off, 32);
    return v;
}

// ---------------------------------------------------------------------------
// Kernel 1: per-cell gene attention (all WMMA work). One 256-thread block
// (8 wave32) per cell. Produces h_all [N,256] and Wh1/Wh2 [N] in workspace.
// ---------------------------------------------------------------------------
__global__ __launch_bounds__(256)
void gene_attn_kernel(const float* __restrict__ h, const int* __restrict__ kadj,
                      const float* __restrict__ Wh_w, const float* __restrict__ Wh_b,
                      const float* __restrict__ Q_w,  const float* __restrict__ Q_b,
                      const float* __restrict__ K_w,  const float* __restrict__ K_b,
                      const float* __restrict__ a_gene, const float* __restrict__ W_cell,
                      const float* __restrict__ a_cell,
                      float* __restrict__ g_hall, float* __restrict__ g_w1,
                      float* __restrict__ g_w2, int N)
{
    __shared__ float  sh_h[F_];
    __shared__ float  sh_hall[2 * F_];
    __shared__ __align__(16) __bf16 sh_Wh[F_ * DK_];
    __shared__ __align__(16) __bf16 sh_Q [F_ * DK_];
    __shared__ __align__(16) __bf16 sh_K [F_ * DK_];
    __shared__ __align__(16) __bf16 sh_KT[F_ * DK_];       // K_cc^T : [f][d]
    __shared__ __align__(16) __bf16 sh_wT[2 * DK_ * DK_];  // Q_w^T then K_w^T
    __shared__ float  sh_bias[2 * DK_];                    // Q_b, K_b
    __shared__ float  sh_lin [2 * DK_];                    // Wh_w, Wh_b
    __shared__ float  sh_S[F_ * F_];                       // 64 KB scores
    __shared__ __align__(16) float sh_hnb[KN_ * F_];       // TDM gather target
    __shared__ float  sh_a[DK_ * KN_];
    __shared__ float  sh_red[4 * EMB_];
#if !USE_TDM
    __shared__ int    sh_nb[KN_];
#endif

    const int n    = blockIdx.x;
    const int t    = threadIdx.x;
    const int lane = t & 31;
    const int wv   = t >> 5;                     // 0..7

#if USE_TDM
    // ---- TDM gather-mode DMA: h[kadj[n,:]] -> sh_hnb, overlapped with WMMA.
    // D# per CDNA5 ISA 8.3-8.7; gather mode, 16-bit indices, 16 rows of
    // tile_dim0=128 x 4B. Issued by wave 0 only (TDM ignores EXEC).
    if (wv == 0) {
        const int* krow = kadj + (size_t)n * KN_;
        unsigned long long ga = (unsigned long long)(const void*)h;
        v4u g0;
        g0[0] = 0x80000001u;                          // count=1, gather_mode=1, 16b idx
        g0[1] = lds_offset(sh_hnb);                   // lds_addr
        g0[2] = (unsigned)(ga & 0xffffffffu);         // global_addr[31:0]
        g0[3] = (unsigned)((ga >> 32) & 0x01ffffffu)  // global_addr[56:32]
              | 0x80000000u;                          // type=2 ("image")
        v8i g1;
        g1[0] = (2 << 16);                            // data_size=4B, wg_mask=0
        g1[1] = (F_ & 0xffff) << 16;                  // tensor_dim0=128 (bits 79:48)
        g1[2] = (N & 0xffff) << 16;                   // tensor_dim1=N   (bits 111:80)
        g1[3] = ((N >> 16) & 0xffff) | (F_ << 16);    // dim1 hi | tile_dim0=128
        g1[4] = KN_;                                  // tile_dim1 = 16 indices
        g1[5] = F_;                                   // tensor_dim0_stride = 128
        g1[6] = 0;
        g1[7] = 0;
        v4i g2, g3;
#pragma unroll
        for (int j = 0; j < 4; ++j) {
            g2[j] = (krow[2 * j]     & 0xffff) | (krow[2 * j + 1] << 16);
            g3[j] = (krow[8 + 2 * j] & 0xffff) | (krow[9 + 2 * j] << 16);
        }
        v8i g4 = {0, 0, 0, 0, 0, 0, 0, 0};            // extra words (zero-filled)
        __builtin_amdgcn_tensor_load_to_lds(g0, g1, g2, g3, g4, 0);
    }
#else
    if (t < KN_) sh_nb[t] = kadj[(size_t)n * KN_ + t];
#endif

    // ---- stage 0: stage inputs into LDS --------------------------------
    for (int i = t; i < F_; i += 256) sh_h[i] = h[(size_t)n * F_ + i];
    for (int i = t; i < DK_ * DK_; i += 256) {   // transpose weights -> bf16
        int k = i >> 5, d = i & 31;
        sh_wT[d * DK_ + k]             = (__bf16)Q_w[i];
        sh_wT[DK_*DK_ + d * DK_ + k]   = (__bf16)K_w[i];
    }
    if (t < DK_)            sh_bias[t]        = Q_b[t];
    else if (t < 2 * DK_)   sh_bias[t]        = K_b[t - DK_];
    if (t >= 64 && t < 64 + DK_) sh_lin[t - 64]         = Wh_w[t - 64];
    if (t >= 96 && t < 96 + DK_) sh_lin[DK_ + (t - 96)] = Wh_b[t - 96];
    for (int i = t; i < DK_ * KN_; i += 256) sh_a[i] = a_gene[i];
    __syncthreads();

#if !USE_TDM
    // fallback gather (L2-resident; coalesced per row)
    for (int i = t; i < KN_ * F_; i += 256) {
        int k = i >> 7, f = i & 127;
        sh_hnb[i] = h[(size_t)sh_nb[k] * F_ + f];
    }
#endif

    // ---- stage 1: Wh = relu(h*w + b) in bf16 ---------------------------
    for (int i = t; i < F_ * DK_; i += 256) {
        int f = i >> 5, d = i & 31;
        float v = sh_h[f] * sh_lin[d] + sh_lin[DK_ + d];
        sh_Wh[i] = (__bf16)fmaxf(v, 0.0f);
    }
    __syncthreads();

    const int col = lane & 15;
    const int mb  = (lane >> 4) << 3;

    // ---- stage 2: Q = Wh@Qw + qb ; K = Wh@Kw + kb (WMMA) ---------------
    {
        const int tm = wv;
        v16bf afrag = ld_frag32(sh_Wh, tm * 16, lane);
#pragma unroll
        for (int tn = 0; tn < 2; ++tn) {
            v16bf bq = ld_frag32(sh_wT,             tn * 16, lane);
            v16bf bk = ld_frag32(sh_wT + DK_ * DK_, tn * 16, lane);
            v8f cq = {}; v8f ck = {};
            cq = __builtin_amdgcn_wmma_f32_16x16x32_bf16(false, afrag, false, bq,
                                                         (short)0, cq, false, false);
            ck = __builtin_amdgcn_wmma_f32_16x16x32_bf16(false, afrag, false, bk,
                                                         (short)0, ck, false, false);
            const int dcol = tn * 16 + col;
            const float qb = sh_bias[dcol], kb = sh_bias[DK_ + dcol];
#pragma unroll
            for (int r = 0; r < 8; ++r) {
                const int row = tm * 16 + mb + r;
                sh_Q[row * DK_ + dcol] = (__bf16)(cq[r] + qb);
                sh_K[row * DK_ + dcol] = (__bf16)(ck[r] + kb);
            }
        }
    }
    __syncthreads();

    // ---- stage 3: S_re = Q K^T / sqrt(dk) (64 WMMA tiles, 8/wave) ------
    {
        const int tm = wv;
        v16bf afrag = ld_frag32(sh_Q, tm * 16, lane);
#pragma unroll
        for (int tn = 0; tn < 8; ++tn) {
            v16bf bfrag = ld_frag32(sh_K, tn * 16, lane);  // B[k][j]=K[tn*16+j][k]
            v8f c = {};
            c = __builtin_amdgcn_wmma_f32_16x16x32_bf16(false, afrag, false, bfrag,
                                                        (short)0, c, false, false);
#pragma unroll
            for (int r = 0; r < 8; ++r)
                sh_S[(tm * 16 + mb + r) * F_ + tn * 16 + col] = c[r] * INV_SCALE;
        }
    }
    __syncthreads();

    // ---- stage 4: row softmax + matvec(h) -> h_all[0:F] ----------------
    for (int rr = 0; rr < 16; ++rr) {
        const int f = wv * 16 + rr;
        const float* Srow = &sh_S[f * F_];
        float s0 = Srow[lane], s1 = Srow[lane + 32], s2 = Srow[lane + 64], s3 = Srow[lane + 96];
        float m  = wave_max(fmaxf(fmaxf(s0, s1), fmaxf(s2, s3)));
        float e0 = __expf(s0 - m), e1 = __expf(s1 - m), e2 = __expf(s2 - m), e3 = __expf(s3 - m);
        float se = wave_sum(e0 + e1 + e2 + e3);
        float sh = wave_sum(e0 * sh_h[lane] + e1 * sh_h[lane + 32] +
                            e2 * sh_h[lane + 64] + e3 * sh_h[lane + 96]);
        if (lane == 0) sh_hall[f] = sh / se + sh_h[f];
    }

#if USE_TDM
    if (wv == 0) __builtin_amdgcn_s_wait_tensorcnt(0);  // sh_hnb ready
#endif
    __syncthreads();

    // ---- stage 5: K_cc^T[f][d] = sum_k a[d][k] * h_nb[k][f] ------------
    for (int i = t; i < F_ * DK_; i += 256) {
        int f = i >> 5, d = i & 31;
        float acc = 0.f;
#pragma unroll
        for (int k = 0; k < KN_; ++k) acc += sh_a[d * KN_ + k] * sh_hnb[k * F_ + f];
        sh_KT[f * DK_ + d] = (__bf16)acc;
    }
    __syncthreads();

    // ---- stage 6: S_cc = Q K_cc / sqrt(dk) -----------------------------
    {
        const int tm = wv;
        v16bf afrag = ld_frag32(sh_Q, tm * 16, lane);
#pragma unroll
        for (int tn = 0; tn < 8; ++tn) {
            v16bf bfrag = ld_frag32(sh_KT, tn * 16, lane);
            v8f c = {};
            c = __builtin_amdgcn_wmma_f32_16x16x32_bf16(false, afrag, false, bfrag,
                                                        (short)0, c, false, false);
#pragma unroll
            for (int r = 0; r < 8; ++r)
                sh_S[(tm * 16 + mb + r) * F_ + tn * 16 + col] = c[r] * INV_SCALE;
        }
    }
    __syncthreads();

    // ---- stage 7: softmax + matvec(h) -> h_all[F:2F] -------------------
    for (int rr = 0; rr < 16; ++rr) {
        const int f = wv * 16 + rr;
        const float* Srow = &sh_S[f * F_];
        float s0 = Srow[lane], s1 = Srow[lane + 32], s2 = Srow[lane + 64], s3 = Srow[lane + 96];
        float m  = wave_max(fmaxf(fmaxf(s0, s1), fmaxf(s2, s3)));
        float e0 = __expf(s0 - m), e1 = __expf(s1 - m), e2 = __expf(s2 - m), e3 = __expf(s3 - m);
        float se = wave_sum(e0 + e1 + e2 + e3);
        float sh = wave_sum(e0 * sh_h[lane] + e1 * sh_h[lane + 32] +
                            e2 * sh_h[lane + 64] + e3 * sh_h[lane + 96]);
        if (lane == 0) sh_hall[F_ + f] = sh / se + sh_h[f];
    }
    __syncthreads();

    // ---- stage 8: Wh_cc = h_all @ W_cell ; Wh1/Wh2 ; write outputs -----
    {
        const int e = t & 63, part = t >> 6;
        float acc = 0.f;
#pragma unroll 4
        for (int j = part * 64; j < part * 64 + 64; ++j)
            acc += sh_hall[j] * W_cell[j * EMB_ + e];
        sh_red[part * EMB_ + e] = acc;
    }
    __syncthreads();
    if (t < EMB_) {
        const float whe = sh_red[t] + sh_red[EMB_ + t] + sh_red[2 * EMB_ + t] + sh_red[3 * EMB_ + t];
        sh_red[t]        = whe * a_cell[t];          // partial for Wh1
        sh_red[EMB_ + t] = whe * a_cell[EMB_ + t];   // partial for Wh2
    }
    __syncthreads();
    if (wv == 0) {
        float p1 = sh_red[lane] + sh_red[lane + 32];
        float p2 = sh_red[EMB_ + lane] + sh_red[EMB_ + lane + 32];
        p1 = wave_sum(p1);
        p2 = wave_sum(p2);
        if (lane == 0) { g_w1[n] = p1; g_w2[n] = p2; }
    }
    g_hall[(size_t)n * 2 * F_ + t] = sh_hall[t];
}

// ---------------------------------------------------------------------------
// Kernel 2: GAT-style neighbor-cell attention + elu. One block per cell;
// thread t owns feature t of the 2F=256-wide output row.
// ---------------------------------------------------------------------------
__global__ __launch_bounds__(256)
void cell_attn_kernel(const float* __restrict__ g_hall, const float* __restrict__ g_w1,
                      const float* __restrict__ g_w2, const int* __restrict__ kadj,
                      float* __restrict__ out)
{
    __shared__ int   nb[KN_];
    __shared__ float sc[KN_];
    const int n = blockIdx.x, t = threadIdx.x;
    if (t < KN_) {
        const int nbk = kadj[(size_t)n * KN_ + t];
        nb[t] = nbk;
        const float s = g_w1[nbk] + g_w2[n];
        sc[t] = s > 0.f ? s : ALPHA_ * s;                         // leaky_relu
        __builtin_prefetch(&g_hall[(size_t)nbk * 2 * F_], 0, 0);  // global_prefetch_b8
    }
    __syncthreads();

    float mx = -INFINITY;
#pragma unroll
    for (int k = 0; k < KN_; ++k) mx = fmaxf(mx, sc[k]);
    float w[KN_], den = 0.f;
#pragma unroll
    for (int k = 0; k < KN_; ++k) { w[k] = __expf(sc[k] - mx); den += w[k]; }
    const float inv = 1.f / den;

    float acc = g_hall[(size_t)n * 2 * F_ + t];
#pragma unroll
    for (int k = 0; k < KN_; ++k)
        acc += (w[k] * inv) * g_hall[(size_t)nb[k] * 2 * F_ + t];

    out[(size_t)n * 2 * F_ + t] = acc > 0.f ? acc : (__expf(acc) - 1.f);  // elu
}

// ---------------------------------------------------------------------------
extern "C" void kernel_launch(void* const* d_in, const int* in_sizes, int n_in,
                              void* d_out, int out_size, void* d_ws, size_t ws_size,
                              hipStream_t stream) {
    const float* h      = (const float*)d_in[0];
    const int*   kadj   = (const int*)  d_in[1];
    const float* Wh_w   = (const float*)d_in[2];
    const float* Wh_b   = (const float*)d_in[3];
    const float* Q_w    = (const float*)d_in[4];
    const float* Q_b    = (const float*)d_in[5];
    const float* K_w    = (const float*)d_in[6];
    const float* K_b    = (const float*)d_in[7];
    const float* a_gene = (const float*)d_in[8];
    const float* W_cell = (const float*)d_in[9];
    const float* a_cell = (const float*)d_in[10];
    const int N = in_sizes[0] / F_;

    float* g_hall = (float*)d_ws;                 // [N, 2F]
    float* g_w1   = g_hall + (size_t)N * 2 * F_;  // [N]
    float* g_w2   = g_w1 + N;                     // [N]

    gene_attn_kernel<<<N, 256, 0, stream>>>(h, kadj, Wh_w, Wh_b, Q_w, Q_b, K_w, K_b,
                                            a_gene, W_cell, a_cell, g_hall, g_w1, g_w2, N);
    cell_attn_kernel<<<N, 256, 0, stream>>>(g_hall, g_w1, g_w2, kadj, (float*)d_out);
}
